// STVimTokenMambaEncoder_21784074125467
// MI455X (gfx1250) — matrix operand order repaired and checked by reading, not compiled, not run on hardware
//
#include <hip/hip_runtime.h>

// ===========================================================================
// STVim Token Mamba Encoder for MI455X (gfx1250, wave32, WMMA).
//
// Shapes: BT=16 frames, 197 tokens -> M=3152; D=384, Di=768, n=16 state,
// k=4 conv, 8 bidirectional spatial layers + 2 temporal layers.
//
// All GEMMs: v_wmma_f32_16x16x32_bf16 (bf16 A/B, f32 accum). Activations that
// feed a GEMM are written in bf16 by their producer kernels, so the GEMM
// inner loop is pure global b128 loads + WMMA (no LDS, no barriers, no cvt).
// Weights converted once to padded bf16 (xproj 56->64 rows, dt_w K 24->32
// zero-padded so stray xdbl columns are annihilated).
// The SSM scan stages its per-step B/C broadcast with the CDNA5 async
// global->LDS pipe (global_load_async_to_lds_b32 + s_wait_asynccnt).
// ===========================================================================

typedef __attribute__((ext_vector_type(16))) __bf16 v16bf;
typedef __attribute__((ext_vector_type(8)))  float  v8f;

#define EW_BLOCK 256
#define GEMM_WAVES 4
#define ASTR 40            // LDS row stride (bf16) for patch-embed A staging

__device__ __forceinline__ float silu_f(float x)     { return x / (1.f + __expf(-x)); }
__device__ __forceinline__ float softplus_f(float x) { return (x > 20.f) ? x : log1pf(__expf(x)); }

union FragU { uint4 q[2]; v16bf v; };

// A fragment (16x32 bf16, MxK), lane holds row r=lane&15.
// Lanes 0-15: K 0..7 (v0..3) + 16..23 (v4..7); lanes 16-31: +8.
// => two contiguous 16B chunks at element offsets hi and 16+hi.
__device__ __forceinline__ v16bf fragA_glb(const __bf16* Arow, int k0, int hi) {
  FragU u;
  u.q[0] = *(const uint4*)(Arow + k0 + hi);
  u.q[1] = *(const uint4*)(Arow + k0 + 16 + hi);
  return u.v;
}

// B fragment (32x16 bf16, KxN), lane holds column n (weight row n, (N,K)
// row-major). Lanes 0-15: K 0..7 + 16..23; lanes 16-31: +8 -> two 16B chunks.
__device__ __forceinline__ v16bf fragB_glb(const __bf16* Brow, int k0, int kh) {
  FragU u;
  u.q[0] = *(const uint4*)(Brow + k0 + kh);
  u.q[1] = *(const uint4*)(Brow + k0 + 16 + kh);
  return u.v;
}

// ---------------------------------------------------------------------------
// GEMM: C[M,N] = act( A[M,K](bf16) @ B[N,K](bf16)^T + bias[N] ), optional
// bf16 mirror of C. One wave computes TN adjacent 16x16 tiles (A reuse).
// grid = (M/16, N/(64*TN)), block = 128 (4 waves).
// ---------------------------------------------------------------------------
template <int TN>
__global__ void gemm_bf16_kernel(const __bf16* __restrict__ A, int lda,
                                 const __bf16* __restrict__ B, int ldb,
                                 float* __restrict__ C, int ldc, int K,
                                 const float* __restrict__ bias, int act,
                                 __bf16* __restrict__ Cbf) {
  const int tid  = threadIdx.x;
  const int wave = tid >> 5, lane = tid & 31;
  const int m0 = blockIdx.x * 16;
  const int n0 = (blockIdx.y * GEMM_WAVES + wave) * 16 * TN;
  const int r = lane & 15, half = lane >> 4;
  const int hiA = half * 8, khB = half * 8;

  v8f acc[TN];
#pragma unroll
  for (int t = 0; t < TN; ++t) acc[t] = (v8f){0.f,0.f,0.f,0.f,0.f,0.f,0.f,0.f};

  const __bf16* Arow = A + (size_t)(m0 + r) * lda;
  const __bf16* Brow[TN];
#pragma unroll
  for (int t = 0; t < TN; ++t) Brow[t] = B + (size_t)(n0 + t * 16 + r) * ldb;

  for (int k0 = 0; k0 < K; k0 += 32) {
    __builtin_prefetch(Arow + k0 + 128, 0, 1);      // global_prefetch_b8
    v16bf a = fragA_glb(Arow, k0, hiA);
#pragma unroll
    for (int t = 0; t < TN; ++t) {
      v16bf b = fragB_glb(Brow[t], k0, khB);
      acc[t] = __builtin_amdgcn_wmma_f32_16x16x32_bf16(false, a, false, b,
                                                       (short)0, acc[t], false, false);
    }
  }

#pragma unroll
  for (int t = 0; t < TN; ++t) {
    const int n = n0 + t * 16 + r;
#pragma unroll
    for (int v = 0; v < 8; ++v) {
      const int m = m0 + v + half * 8;
      float x = acc[t][v];
      if (bias) x += bias[n];
      if (act == 1) x = softplus_f(x);
      C[(size_t)m * ldc + n] = x;
      if (Cbf) Cbf[(size_t)m * ldc + n] = (__bf16)x;
    }
  }
}

// ---------------------------------------------------------------------------
// Patch-embed GEMM: A gathered from frames (16,1,224,224) + f32->bf16 into
// LDS. M=3136, K=256, N=384.  C = patches @ Wpe^T (f32 out).
// ---------------------------------------------------------------------------
__global__ void patch_gemm_kernel(const float* __restrict__ frames,
                                  const __bf16* __restrict__ B,   // Wpe bf16 384x256
                                  float* __restrict__ C, int ldc) {
  __shared__ __bf16 As[16 * ASTR];
  const int tid  = threadIdx.x;
  const int wave = tid >> 5, lane = tid & 31;
  const int m0 = blockIdx.x * 16;
  const int n0 = (blockIdx.y * GEMM_WAVES + wave) * 16;
  const int r = lane & 15, half = lane >> 4;
  const int hiA = half * 8, khB = half * 8;

  v8f acc = {0.f, 0.f, 0.f, 0.f, 0.f, 0.f, 0.f, 0.f};
  const __bf16* Brow = B + (size_t)(n0 + r) * 256;

  const int arow = tid >> 3;
  const int acol = (tid & 7) * 4;
  const int m = m0 + arow;
  const int fr = m / 196, t = m % 196;
  const int pr = t / 14, pc = t % 14;
  const float* fbase = frames + (size_t)fr * 224 * 224;

  for (int k0 = 0; k0 < 256; k0 += 32) {
    __bf16* d = As + arow * ASTR + acol;
#pragma unroll
    for (int q = 0; q < 4; ++q) {
      const int k = k0 + acol + q;
      const int pi = k >> 4, pj = k & 15;
      d[q] = (__bf16)fbase[(pr * 16 + pi) * 224 + (pc * 16 + pj)];
    }
    __syncthreads();
    v16bf a;
    {
      const __bf16* row = As + r * ASTR;
#pragma unroll
      for (int j = 0; j < 8; ++j) { a[j] = row[hiA + j]; a[8 + j] = row[16 + hiA + j]; }
    }
    v16bf b = fragB_glb(Brow, k0, khB);
    acc = __builtin_amdgcn_wmma_f32_16x16x32_bf16(false, a, false, b,
                                                  (short)0, acc, false, false);
    __syncthreads();
  }

  const int n = n0 + r;
#pragma unroll
  for (int v = 0; v < 8; ++v) {
    const int mo = m0 + v + half * 8;
    C[(size_t)mo * ldc + n] = acc[v];
  }
}

// f32 -> bf16 weight conversion with zero padding to (Np, Kp).
__global__ void cvt_bf16_pad_kernel(const float* __restrict__ src, __bf16* __restrict__ dst,
                                    int N, int K, int Kp, int total) {
  const int i = blockIdx.x * blockDim.x + threadIdx.x;
  if (i >= total) return;
  const int n = i / Kp, k = i % Kp;
  const float v = (n < N && k < K) ? src[(size_t)n * K + k] : 0.f;
  dst[i] = (__bf16)v;
}

// ---------------------------------------------------------------------------
// residual = first ? h : h + residual ;  norm = rmsnorm(residual) * w
// writes f32 (outf) and/or bf16 (outbf).  One block (128 thr) per token.
// ---------------------------------------------------------------------------
__global__ void resid_rmsnorm_kernel(const float* __restrict__ h, float* __restrict__ residual,
                                     const float* __restrict__ w,
                                     float* outf, __bf16* outbf,
                                     int D, int first) {
  __shared__ float red[4];
  const size_t row = blockIdx.x;
  const float* hr = h + row * D;
  float* rr = residual + row * D;
  float ss = 0.f;
  for (int d = threadIdx.x; d < D; d += blockDim.x) {
    float v = hr[d];
    if (!first) v += rr[d];
    rr[d] = v;
    ss += v * v;
  }
#pragma unroll
  for (int off = 16; off > 0; off >>= 1) ss += __shfl_xor(ss, off);
  if ((threadIdx.x & 31) == 0) red[threadIdx.x >> 5] = ss;
  __syncthreads();
  if (threadIdx.x == 0) {
    float t = red[0] + red[1] + red[2] + red[3];
    red[0] = rsqrtf(t / (float)D + 1e-5f);
  }
  __syncthreads();
  const float s = red[0];
  for (int d = threadIdx.x; d < D; d += blockDim.x) {
    const float v = rr[d] * s * w[d];
    if (outf)  outf[row * D + d] = v;
    if (outbf) outbf[row * D + d] = (__bf16)v;
  }
}

// Assemble tokens: insert cls at 98, add bpe (patch tokens) and pos.
__global__ void assemble_tokens_kernel(const float* __restrict__ pe,   // 3136 x 384
                                       const float* __restrict__ bpe,
                                       const float* __restrict__ cls,
                                       const float* __restrict__ pos,  // 197 x 384
                                       float* __restrict__ x) {
  const int i = blockIdx.x * blockDim.x + threadIdx.x;
  const int total = 16 * 197 * 384;
  if (i >= total) return;
  const int d = i % 384;
  const int j = (i / 384) % 197;
  const int f = i / (384 * 197);
  float v;
  if (j == 98) v = cls[d];
  else {
    const int jj = (j < 98) ? j : j - 1;
    v = pe[((size_t)f * 196 + jj) * 384 + d] + bpe[d];
  }
  x[i] = v + pos[j * 384 + d];
}

// ---------------------------------------------------------------------------
// Causal conv1d (k=4) + silu; rev=1 reads the sequence reversed (backward
// branch). Writes f32 (for the scan) and bf16 (GEMM A) in scan order.
// ---------------------------------------------------------------------------
__global__ void conv_silu_kernel(const float* __restrict__ xx, int ldx,
                                 const float* __restrict__ cw,  // Di x 4
                                 const float* __restrict__ cb,  // Di
                                 float* __restrict__ xc, __bf16* __restrict__ xcbf,
                                 int BT, int Lt, int Di, int rev) {
  const int i = blockIdx.x * blockDim.x + threadIdx.x;
  const int total = BT * Lt * Di;
  if (i >= total) return;
  const int d = i % Di;
  const int t = (i / Di) % Lt;
  const int f = i / (Di * Lt);
  float acc = cb[d];
#pragma unroll
  for (int q = 0; q < 4; ++q) {
    const int ts = t - 3 + q;
    if (ts < 0) continue;
    const int src = rev ? (Lt - 1 - ts) : ts;
    acc += cw[d * 4 + q] * xx[((size_t)f * Lt + src) * ldx + d];
  }
  const float v = silu_f(acc);
  const size_t o = ((size_t)f * Lt + t) * Di + d;
  xc[o] = v;
  xcbf[o] = (__bf16)v;
}

// ---------------------------------------------------------------------------
// Selective-scan: one thread per (frame, channel d); h[16] in registers.
// Per step, the 32 contiguous floats xdbl[row, 24..55] (B then C) are staged
// into LDS with the CDNA5 async global->LDS pipe (one lane-parallel
// global_load_async_to_lds_b32, ASYNCcnt-synchronized).
// y may alias dtv (read-before-write per element).
// ---------------------------------------------------------------------------
__global__ void ssm_scan_kernel(const float* __restrict__ xc,
                                const float* dtv,
                                const float* __restrict__ xdbl,  // M x 64: B@24, C@40
                                const float* __restrict__ A_log, // Di x 16
                                const float* __restrict__ Dp,    // Di
                                float* y,
                                int Lt, int Di) {
  __shared__ float sBC[32];          // [0..15] = B, [16..31] = C
  const int f = blockIdx.y;
  const int tid = threadIdx.x;
  const int d = blockIdx.x * blockDim.x + tid;
  float A[16], h[16];
#pragma unroll
  for (int n = 0; n < 16; ++n) { A[n] = -__expf(A_log[(size_t)d * 16 + n]); h[n] = 0.f; }
  const float Dpd = Dp[d];

  // Per-lane LDS byte address for the async copy (flat low 32 bits == LDS addr).
  const unsigned lds_addr = (unsigned)(size_t)(&sBC[tid & 31]);

  for (int t = 0; t < Lt; ++t) {
    const size_t row = (size_t)f * Lt + t;
    __syncthreads();                 // previous step's sBC reads are done
    if (tid < 32) {
      const float* g = xdbl + row * 64 + 24 + tid;   // cols 24..55 contiguous
      asm volatile("global_load_async_to_lds_b32 %0, %1, off"
                   :: "v"(lds_addr), "v"(g) : "memory");
    }
    asm volatile("s_wait_asynccnt 0" ::: "memory");
    __syncthreads();                 // LDS visible to all waves

    const float u   = xc[row * Di + d];
    const float dtt = dtv[row * Di + d];
    float yv = 0.f;
#pragma unroll
    for (int n = 0; n < 16; ++n) {
      const float wgt = __expf(dtt * A[n]);
      h[n] = wgt * h[n] + (dtt * sBC[n]) * u;
      yv += h[n] * sBC[16 + n];
    }
    y[row * Di + d] = yv + Dpd * u;
  }
}

// g = (yf + reverse(yb)) * silu(z)  -> bf16 (GEMM-A only consumer)
__global__ void combine_gate_kernel(const float* __restrict__ yf, const float* yb,
                                    const float* __restrict__ z, int ldz,
                                    __bf16* __restrict__ g,
                                    int BT, int Lt, int Di) {
  const int i = blockIdx.x * blockDim.x + threadIdx.x;
  const int total = BT * Lt * Di;
  if (i >= total) return;
  const int d = i % Di;
  const int t = (i / Di) % Lt;
  const int f = i / (Di * Lt);
  const size_t row = (size_t)f * Lt + t;
  float v = yf[row * Di + d];
  if (yb) v += yb[((size_t)f * Lt + (Lt - 1 - t)) * Di + d];
  g[row * Di + d] = (__bf16)(v * silu_f(z[row * ldz + d]));
}

// hf: 3152 x 384 -> ft (16 x 384), f32 + bf16
__global__ void extract_cls_kernel(const float* __restrict__ hf,
                                   float* __restrict__ ft, __bf16* __restrict__ ftbf) {
  const int i = blockIdx.x * blockDim.x + threadIdx.x;
  if (i >= 16 * 384) return;
  const int d = i % 384, fr = i / 384;
  const float v = hf[((size_t)fr * 197 + 98) * 384 + d];
  ft[i] = v;
  ftbf[i] = (__bf16)v;
}

// out[b,d] = ft[b*8+7, d]
__global__ void final_out_kernel(const float* __restrict__ ft, float* __restrict__ out) {
  const int i = blockIdx.x * blockDim.x + threadIdx.x;
  if (i >= 2 * 384) return;
  const int d = i % 384, b = i / 384;
  out[i] = ft[((size_t)(b * 8 + 7)) * 384 + d];
}

// ===========================================================================
// Host orchestration
// ===========================================================================
extern "C" void kernel_launch(void* const* d_in, const int* in_sizes, int n_in,
                              void* d_out, int out_size, void* d_ws, size_t ws_size,
                              hipStream_t stream) {
  (void)in_sizes; (void)n_in; (void)out_size; (void)ws_size;

  const float* depth_seq = (const float*)d_in[0];
  const float* Wpe       = (const float*)d_in[2];
  const float* bpe       = (const float*)d_in[3];
  const float* cls       = (const float*)d_in[4];
  const float* pos       = (const float*)d_in[5];
  const float* norm_w    = (const float*)d_in[6];
  const float* norm_f_w  = (const float*)d_in[7];
  const float* in_w      = (const float*)d_in[8];
  const float* out_w     = (const float*)d_in[9];
  const float* conv_w    = (const float*)d_in[10];
  const float* conv_b    = (const float*)d_in[11];
  const float* xproj_w   = (const float*)d_in[12];
  const float* dt_w      = (const float*)d_in[13];
  const float* dt_b      = (const float*)d_in[14];
  const float* A_log     = (const float*)d_in[15];
  const float* Dp        = (const float*)d_in[16];
  const float* conv_w_b  = (const float*)d_in[17];
  const float* conv_b_b  = (const float*)d_in[18];
  const float* xproj_w_b = (const float*)d_in[19];
  const float* dt_w_b    = (const float*)d_in[20];
  const float* dt_b_b    = (const float*)d_in[21];
  const float* A_log_b   = (const float*)d_in[22];
  const float* Dp_b      = (const float*)d_in[23];
  const float* t_in_w    = (const float*)d_in[24];
  const float* t_out_w   = (const float*)d_in[25];
  const float* t_conv_w  = (const float*)d_in[26];
  const float* t_conv_b  = (const float*)d_in[27];
  const float* t_xproj_w = (const float*)d_in[28];
  const float* t_dt_w    = (const float*)d_in[29];
  const float* t_dt_b    = (const float*)d_in[30];
  const float* t_A_log   = (const float*)d_in[31];
  const float* t_Dp      = (const float*)d_in[32];

  const int BT = 16, Lt = 197, D = 384, Di = 768, NL = 8, NT = 2;
  const int M = BT * Lt;      // 3152
  const int Mpe = 3136;

  // ---- workspace bump allocator (256B aligned) ----
  char* p = (char*)d_ws;
  auto alloc_f = [&](size_t n) -> float* {
    float* r = (float*)p; p += ((n * 4) + 255) & ~(size_t)255; return r;
  };
  auto alloc_b = [&](size_t n) -> __bf16* {
    __bf16* r = (__bf16*)p; p += ((n * 2) + 255) & ~(size_t)255; return r;
  };

  // bf16 weights (padded)
  __bf16* wpe_bf  = alloc_b((size_t)384 * 256);
  __bf16* inw_bf  = alloc_b((size_t)NL * 1536 * 384);
  __bf16* outw_bf = alloc_b((size_t)NL * 384 * 768);
  __bf16* xpf_bf  = alloc_b((size_t)NL * 64 * 768);
  __bf16* dtf_bf  = alloc_b((size_t)NL * 768 * 32);
  __bf16* xpb_bf  = alloc_b((size_t)NL * 64 * 768);
  __bf16* dtb_bf  = alloc_b((size_t)NL * 768 * 32);
  __bf16* tin_bf  = alloc_b((size_t)NT * 1536 * 384);
  __bf16* tout_bf = alloc_b((size_t)NT * 384 * 768);
  __bf16* txp_bf  = alloc_b((size_t)NT * 64 * 768);
  __bf16* tdt_bf  = alloc_b((size_t)NT * 768 * 32);

  // f32 activations
  float* pe_tmp = alloc_f((size_t)Mpe * D);
  float* hbuf   = alloc_f((size_t)M * D);
  float* resid  = alloc_f((size_t)M * D);
  float* hn     = alloc_f((size_t)M * D);      // final norm (f32, for extract)
  float* xz     = alloc_f((size_t)M * 1536);
  float* xcf    = alloc_f((size_t)M * Di);
  float* xcb    = alloc_f((size_t)M * Di);
  float* xdblf  = alloc_f((size_t)M * 64);
  float* xdblb  = alloc_f((size_t)M * 64);
  float* dtfv   = alloc_f((size_t)M * Di);     // dt, then y in-place (fwd)
  float* dtbv   = alloc_f((size_t)M * Di);     // dt, then y in-place (bwd)
  float* ftA    = alloc_f((size_t)16 * D);
  float* ftB    = alloc_f((size_t)16 * D);

  // bf16 activations (GEMM-A operands)
  __bf16* hn_bf  = alloc_b((size_t)M * D);
  __bf16* xcf_bf = alloc_b((size_t)M * Di);
  __bf16* xcb_bf = alloc_b((size_t)M * Di);
  __bf16* xdf_bf = alloc_b((size_t)M * 64);
  __bf16* xdb_bf = alloc_b((size_t)M * 64);
  __bf16* yg_bf  = alloc_b((size_t)M * Di);
  __bf16* ftA_bf = alloc_b((size_t)16 * D);
  __bf16* ftB_bf = alloc_b((size_t)16 * D);

  auto cvt = [&](const float* s, __bf16* d, int N, int K, int Np, int Kp) {
    const int tot = Np * Kp;
    cvt_bf16_pad_kernel<<<(tot + EW_BLOCK - 1) / EW_BLOCK, EW_BLOCK, 0, stream>>>(
        s, d, N, K, Kp, tot);
  };
  auto gemm = [&](const __bf16* A, int lda, const __bf16* B, int ldb,
                  float* C, int ldc, int Mr, int Nr, int Kr,
                  const float* bias, int act, __bf16* Cbf) {
    if (Nr % 128 == 0)
      gemm_bf16_kernel<2><<<dim3(Mr / 16, Nr / 128), dim3(128), 0, stream>>>(
          A, lda, B, ldb, C, ldc, Kr, bias, act, Cbf);
    else
      gemm_bf16_kernel<1><<<dim3(Mr / 16, Nr / 64), dim3(128), 0, stream>>>(
          A, lda, B, ldb, C, ldc, Kr, bias, act, Cbf);
  };
  auto ew_grid = [&](int tot) { return dim3((tot + EW_BLOCK - 1) / EW_BLOCK); };

  // ---- weight conversion ----
  cvt(Wpe, wpe_bf, 384, 256, 384, 256);
  for (int i = 0; i < NL; ++i) {
    cvt(in_w     + (size_t)i * 1536 * 384, inw_bf  + (size_t)i * 1536 * 384, 1536, 384, 1536, 384);
    cvt(out_w    + (size_t)i * 384 * 768,  outw_bf + (size_t)i * 384 * 768,  384, 768, 384, 768);
    cvt(xproj_w  + (size_t)i * 56 * 768,   xpf_bf  + (size_t)i * 64 * 768,   56, 768, 64, 768);
    cvt(dt_w     + (size_t)i * 768 * 24,   dtf_bf  + (size_t)i * 768 * 32,   768, 24, 768, 32);
    cvt(xproj_w_b+ (size_t)i * 56 * 768,   xpb_bf  + (size_t)i * 64 * 768,   56, 768, 64, 768);
    cvt(dt_w_b   + (size_t)i * 768 * 24,   dtb_bf  + (size_t)i * 768 * 32,   768, 24, 768, 32);
  }
  for (int j = 0; j < NT; ++j) {
    cvt(t_in_w   + (size_t)j * 1536 * 384, tin_bf  + (size_t)j * 1536 * 384, 1536, 384, 1536, 384);
    cvt(t_out_w  + (size_t)j * 384 * 768,  tout_bf + (size_t)j * 384 * 768,  384, 768, 384, 768);
    cvt(t_xproj_w+ (size_t)j * 56 * 768,   txp_bf  + (size_t)j * 64 * 768,   56, 768, 64, 768);
    cvt(t_dt_w   + (size_t)j * 768 * 24,   tdt_bf  + (size_t)j * 768 * 32,   768, 24, 768, 32);
  }

  // ---- patch embed + token assembly ----
  patch_gemm_kernel<<<dim3(Mpe / 16, D / 64), dim3(128), 0, stream>>>(
      depth_seq, wpe_bf, pe_tmp, D);
  assemble_tokens_kernel<<<ew_grid(M * D), EW_BLOCK, 0, stream>>>(
      pe_tmp, bpe, cls, pos, hbuf);

  // ---- spatial Mamba layers (bidirectional) ----
  for (int i = 0; i < NL; ++i) {
    resid_rmsnorm_kernel<<<M, 128, 0, stream>>>(
        hbuf, resid, norm_w + (size_t)i * D, nullptr, hn_bf, D, i == 0);

    // xz = hn @ in_w^T   (M x 1536)
    gemm(hn_bf, D, inw_bf + (size_t)i * 1536 * 384, 384, xz, 1536, M, 1536, 384,
         nullptr, 0, nullptr);

    // forward branch
    conv_silu_kernel<<<ew_grid(M * Di), EW_BLOCK, 0, stream>>>(
        xz, 1536, conv_w + (size_t)i * Di * 4, conv_b + (size_t)i * Di,
        xcf, xcf_bf, BT, Lt, Di, 0);
    gemm(xcf_bf, Di, xpf_bf + (size_t)i * 64 * 768, 768, xdblf, 64, M, 64, 768,
         nullptr, 0, xdf_bf);
    gemm(xdf_bf, 64, dtf_bf + (size_t)i * 768 * 32, 32, dtfv, Di, M, Di, 32,
         dt_b + (size_t)i * Di, 1, nullptr);
    ssm_scan_kernel<<<dim3(Di / 256, BT), 256, 0, stream>>>(
        xcf, dtfv, xdblf, A_log + (size_t)i * Di * 16, Dp + (size_t)i * Di, dtfv, Lt, Di);

    // backward branch (sequence reversed per frame)
    conv_silu_kernel<<<ew_grid(M * Di), EW_BLOCK, 0, stream>>>(
        xz, 1536, conv_w_b + (size_t)i * Di * 4, conv_b_b + (size_t)i * Di,
        xcb, xcb_bf, BT, Lt, Di, 1);
    gemm(xcb_bf, Di, xpb_bf + (size_t)i * 64 * 768, 768, xdblb, 64, M, 64, 768,
         nullptr, 0, xdb_bf);
    gemm(xdb_bf, 64, dtb_bf + (size_t)i * 768 * 32, 32, dtbv, Di, M, Di, 32,
         dt_b_b + (size_t)i * Di, 1, nullptr);
    ssm_scan_kernel<<<dim3(Di / 256, BT), 256, 0, stream>>>(
        xcb, dtbv, xdblb, A_log_b + (size_t)i * Di * 16, Dp_b + (size_t)i * Di, dtbv, Lt, Di);

    // gate and out-projection
    combine_gate_kernel<<<ew_grid(M * Di), EW_BLOCK, 0, stream>>>(
        dtfv, dtbv, xz + 768, 1536, yg_bf, BT, Lt, Di);
    gemm(yg_bf, Di, outw_bf + (size_t)i * 384 * 768, 768, hbuf, D, M, D, 768,
         nullptr, 0, nullptr);
  }

  // ---- final norm + frame-token extraction ----
  resid_rmsnorm_kernel<<<M, 128, 0, stream>>>(hbuf, resid, norm_f_w, hn, nullptr, D, 0);
  extract_cls_kernel<<<ew_grid(16 * D), EW_BLOCK, 0, stream>>>(hn, ftA, ftA_bf);

  // ---- temporal Mamba layers (unidirectional, B=2, T=8) ----
  float*   ftin_f  = ftA;  float*   ftout_f  = ftB;
  __bf16*  ftin_b  = ftA_bf; __bf16* ftout_b = ftB_bf;
  for (int j = 0; j < NT; ++j) {
    gemm(ftin_b, D, tin_bf + (size_t)j * 1536 * 384, 384, xz, 1536, 16, 1536, 384,
         nullptr, 0, nullptr);
    conv_silu_kernel<<<ew_grid(16 * Di), EW_BLOCK, 0, stream>>>(
        xz, 1536, t_conv_w + (size_t)j * Di * 4, t_conv_b + (size_t)j * Di,
        xcf, xcf_bf, 2, 8, Di, 0);
    gemm(xcf_bf, Di, txp_bf + (size_t)j * 64 * 768, 768, xdblf, 64, 16, 64, 768,
         nullptr, 0, xdf_bf);
    gemm(xdf_bf, 64, tdt_bf + (size_t)j * 768 * 32, 32, dtfv, Di, 16, Di, 32,
         t_dt_b + (size_t)j * Di, 1, nullptr);
    ssm_scan_kernel<<<dim3(Di / 256, 2), 256, 0, stream>>>(
        xcf, dtfv, xdblf, t_A_log + (size_t)j * Di * 16, t_Dp + (size_t)j * Di, dtfv, 8, Di);
    combine_gate_kernel<<<ew_grid(16 * Di), EW_BLOCK, 0, stream>>>(
        dtfv, nullptr, xz + 768, 1536, yg_bf, 2, 8, Di);
    gemm(yg_bf, Di, tout_bf + (size_t)j * 384 * 768, 768, ftout_f, D, 16, D, 768,
         nullptr, 0, ftout_b);
    { float* tf = ftin_f; ftin_f = ftout_f; ftout_f = tf; }
    { __bf16* tb = ftin_b; ftin_b = ftout_b; ftout_b = tb; }
  }

  final_out_kernel<<<ew_grid(2 * 384), EW_BLOCK, 0, stream>>>(ftin_f, (float*)d_out);
}
